// VectorQuantizer_27513560498892
// MI455X (gfx1250) — compile-verified, matrix operand order checked by reading, not run on hardware
//
#include <hip/hip_runtime.h>
#include <hip/hip_bf16.h>
#include <stdint.h>

#define KCODES 1024
#define DDIM   64
#define NROWS  (64 * 4096)            // 262144 rows
#define Z_ELEMS (NROWS * DDIM)        // 16777216
#define LOSS_OFF Z_ELEMS
#define IDX_OFF  (Z_ELEMS + 1)

#define WAVES   4                     // waves per block (128 threads)
#define RPW     32                    // rows per wave (two 16-row WMMA tiles)
#define NBLKS   (NROWS / (WAVES * RPW)) // 2048 blocks
#define NTILES  (KCODES / 16)         // 64 column tiles
#define ASTRIDE 68                    // padded LDS row stride (floats)

// workspace layout (uint32 units)
#define WS_BHI  0                     // 64*2*32*8 = 32768 u32 (bf16 hi pairs)
#define WS_BLO  32768                 // 32768 u32 (bf16 lo pairs)
#define WS_E2   65536                 // 1024 f32
#define WS_PART 66560                 // 2048 f32 block partial losses
#define BLO_OFF 4096                  // Blo - Bhi distance in v8u units

typedef __attribute__((ext_vector_type(16))) __bf16        v16bf;
typedef __attribute__((ext_vector_type(8)))  float         v8f;
typedef __attribute__((ext_vector_type(8)))  unsigned int  v8u;

// Split f32 into bf16 hi (truncate) + bf16 lo (residual), pack two values per u32.
__device__ __forceinline__ void pack_hi_lo(float x0, float x1,
                                           unsigned int& hi, unsigned int& lo) {
  unsigned int u0 = __float_as_uint(x0), u1 = __float_as_uint(x1);
  unsigned int h0 = u0 & 0xFFFF0000u,   h1 = u1 & 0xFFFF0000u;
  float r0 = x0 - __uint_as_float(h0);
  float r1 = x1 - __uint_as_float(h1);
  hi = (h0 >> 16) | h1;
  lo = (__float_as_uint(r0) >> 16) | (__float_as_uint(r1) & 0xFFFF0000u);
}

__device__ __forceinline__ v8f wmma_bf16(v8u a, v8u b, v8f c) {
  // 8 args: (neg_a, A, neg_b, B, c_mod, C, reuse_a, reuse_b)
  return __builtin_amdgcn_wmma_f32_16x16x32_bf16(
      false, __builtin_bit_cast(v16bf, a),
      false, __builtin_bit_cast(v16bf, b),
      (short)0, c, false, false);
}

// -------- prep: embedding -> WMMA B-fragment-ordered bf16 hi/lo planes --------
// One thread per packed u32: gid = (((t*2+s)*32)+lane)*8 + j2
__global__ void vq_prep_frags(const float* __restrict__ E,
                              unsigned int* __restrict__ ws) {
  int gid = blockIdx.x * blockDim.x + threadIdx.x;   // 0..32767
  int j2   = gid & 7;
  int lane = (gid >> 3) & 31;
  int s    = (gid >> 8) & 1;
  int t    = gid >> 9;
  int half = lane >> 4;
  int code = t * 16 + (lane & 15);
  int j0 = 2 * j2, j1 = j0 + 1;
  // 16-bit fragment K striping (CDNA5 ISA): d_local = j + (j>=8 ? 8:0) + half*8
  int d0 = s * 32 + j0 + ((j0 & 8) ? 8 : 0) + half * 8;
  int d1 = s * 32 + j1 + ((j1 & 8) ? 8 : 0) + half * 8;
  unsigned int hi, lo;
  pack_hi_lo(E[code * DDIM + d0], E[code * DDIM + d1], hi, lo);
  ws[WS_BHI + gid] = hi;
  ws[WS_BLO + gid] = lo;
}

__global__ void vq_prep_e2(const float* __restrict__ E, float* __restrict__ e2) {
  int k = blockIdx.x * blockDim.x + threadIdx.x;
  if (k < KCODES) {
    const float4* row = (const float4*)(E + k * DDIM);
    float s = 0.f;
#pragma unroll
    for (int i = 0; i < DDIM / 4; ++i) {
      float4 v = row[i];
      s = fmaf(v.x, v.x, s); s = fmaf(v.y, v.y, s);
      s = fmaf(v.z, v.z, s); s = fmaf(v.w, v.w, s);
    }
    e2[k] = s;
  }
}

// -------- main: distances via bf16x3 WMMA (2 row-tiles/wave), argmin, gather --------
__global__ void __launch_bounds__(WAVES * 32)
vq_main(const float* __restrict__ z, const float* __restrict__ E,
        const unsigned int* __restrict__ ws, float* __restrict__ out) {
  __shared__ float ldsA[WAVES * RPW * ASTRIDE];  // staged z tiles (fp32)
  __shared__ float ldsE2[KCODES];
  __shared__ int   ldsIdx[WAVES * RPW];
  __shared__ float ldsLoss[WAVES];

  const int tid  = threadIdx.x;
  const int wave = tid >> 5;
  const int lane = tid & 31;
  const int half = lane >> 4;
  const int lrow = lane & 15;
  const int rowBase = (blockIdx.x * WAVES + wave) * RPW;

  // preload e2 into LDS
  for (int i = tid; i < KCODES; i += WAVES * 32)
    ldsE2[i] = __uint_as_float(ws[WS_E2 + i]);

  // stage this wave's 32x64 z tile into LDS (coalesced b128)
  float* A = &ldsA[wave * RPW * ASTRIDE];
  const float4* zsrc = (const float4*)(z + (size_t)rowBase * DDIM);
#pragma unroll
  for (int q = 0; q < RPW; ++q) {            // 512 float4 / 32 lanes = 16/lane
    int i = q * 32 + lane;
    if (q < 16) {
      float4 v = zsrc[i];
      int r = i >> 4, c4 = i & 15;
      *(float4*)&A[r * ASTRIDE + c4 * 4] = v;
    }
  }
  __syncthreads();

  // build A fragments (hi/lo bf16) for both 16-row tiles; reused across 64 B tiles
  v8u ahi[2][2], alo[2][2];
#pragma unroll
  for (int rt = 0; rt < 2; ++rt) {
#pragma unroll
    for (int s = 0; s < 2; ++s) {
      const float* base = A + (rt * 16 + lrow) * ASTRIDE + s * 32 + half * 8;
      float4 c0 = *(const float4*)(base + 0);
      float4 c1 = *(const float4*)(base + 4);
      float4 c2 = *(const float4*)(base + 16);
      float4 c3 = *(const float4*)(base + 20);
      unsigned int h[8], l[8];
      pack_hi_lo(c0.x, c0.y, h[0], l[0]); pack_hi_lo(c0.z, c0.w, h[1], l[1]);
      pack_hi_lo(c1.x, c1.y, h[2], l[2]); pack_hi_lo(c1.z, c1.w, h[3], l[3]);
      pack_hi_lo(c2.x, c2.y, h[4], l[4]); pack_hi_lo(c2.z, c2.w, h[5], l[5]);
      pack_hi_lo(c3.x, c3.y, h[6], l[6]); pack_hi_lo(c3.z, c3.w, h[7], l[7]);
      v8u th = {h[0], h[1], h[2], h[3], h[4], h[5], h[6], h[7]};
      v8u tl = {l[0], l[1], l[2], l[3], l[4], l[5], l[6], l[7]};
      ahi[rt][s] = th; alo[rt][s] = tl;
    }
  }

  const v8u* pb = (const v8u*)(ws + WS_BHI) + lane;   // bumped by 64/tile

  float bestV[2][8]; int bestI[2][8];
#pragma unroll
  for (int rt = 0; rt < 2; ++rt)
#pragma unroll
    for (int j = 0; j < 8; ++j) { bestV[rt][j] = 3.4e38f; bestI[rt][j] = 0; }

  float e2c = ldsE2[lrow];                             // tile 0's e2, pipelined
  for (int t = 0; t < NTILES; ++t) {
    v8u bh0 = pb[0];
    v8u bh1 = pb[32];
    v8u bl0 = pb[BLO_OFF];
    v8u bl1 = pb[BLO_OFF + 32];
    __builtin_prefetch(&pb[64], 0, 3);                 // next tile -> near cache
    __builtin_prefetch(&pb[BLO_OFF + 64], 0, 3);
    float e2n = (t + 1 < NTILES) ? ldsE2[(t + 1) * 16 + lrow] : 0.f;

    // S = Ahi*Bhi + Ahi*Blo + Alo*Bhi  (bf16x3 ~ fp32-accurate dot products),
    // two independent accumulator chains interleave on the XDL pipe
    v8f c0 = {}, c1 = {};
    c0 = wmma_bf16(ahi[0][0], bh0, c0);  c1 = wmma_bf16(ahi[1][0], bh0, c1);
    c0 = wmma_bf16(ahi[0][1], bh1, c0);  c1 = wmma_bf16(ahi[1][1], bh1, c1);
    c0 = wmma_bf16(ahi[0][0], bl0, c0);  c1 = wmma_bf16(ahi[1][0], bl0, c1);
    c0 = wmma_bf16(ahi[0][1], bl1, c0);  c1 = wmma_bf16(ahi[1][1], bl1, c1);
    c0 = wmma_bf16(alo[0][0], bh0, c0);  c1 = wmma_bf16(alo[1][0], bh0, c1);
    c0 = wmma_bf16(alo[0][1], bh1, c0);  c1 = wmma_bf16(alo[1][1], bh1, c1);

    int idx = t * 16 + lrow;
#pragma unroll
    for (int j = 0; j < 8; ++j) {
      float v0 = fmaf(-2.0f, c0[j], e2c);   // |x|^2 is row-constant: dropped
      float v1 = fmaf(-2.0f, c1[j], e2c);
      if (v0 < bestV[0][j]) { bestV[0][j] = v0; bestI[0][j] = idx; }
      if (v1 < bestV[1][j]) { bestV[1][j] = v1; bestI[1][j] = idx; }
    }
    e2c = e2n;
    pb += 64;
  }

  // cross-lane argmin within each 16-lane half (row r lives in slot r / r-8)
#pragma unroll
  for (int rt = 0; rt < 2; ++rt) {
#pragma unroll
    for (int j = 0; j < 8; ++j) {
      float v = bestV[rt][j]; int i = bestI[rt][j];
#pragma unroll
      for (int m = 1; m <= 8; m <<= 1) {
        float ov = __shfl_xor(v, m, 32);
        int   oi = __shfl_xor(i, m, 32);
        if (ov < v || (ov == v && oi < i)) { v = ov; i = oi; } // first-min tie-break
      }
      bestV[rt][j] = v; bestI[rt][j] = i;
    }
  }

  if (lrow == 0) {
#pragma unroll
    for (int rt = 0; rt < 2; ++rt)
#pragma unroll
      for (int j = 0; j < 8; ++j) {
        int r = rt * 16 + half * 8 + j;
        out[IDX_OFF + rowBase + r] = (float)bestI[rt][j];
        ldsIdx[wave * RPW + r] = bestI[rt][j];
      }
  }
  __syncthreads();

  // gather z_q (exact fp32 from embedding), write out, accumulate loss
  float loss = 0.f;
#pragma unroll 4
  for (int r = 0; r < RPW; ++r) {
    int idx = ldsIdx[wave * RPW + r];
    float2 e = *(const float2*)(E + idx * DDIM + lane * 2);
    float zx = A[r * ASTRIDE + lane * 2];
    float zy = A[r * ASTRIDE + lane * 2 + 1];
    *(float2*)&out[(size_t)(rowBase + r) * DDIM + lane * 2] = e;
    float dx = e.x - zx, dy = e.y - zy;
    loss = fmaf(dx, dx, loss);
    loss = fmaf(dy, dy, loss);
  }
#pragma unroll
  for (int m = 1; m < 32; m <<= 1) loss += __shfl_xor(loss, m, 32);
  if (lane == 0) ldsLoss[wave] = loss;
  __syncthreads();
  if (tid == 0) {
    float s = 0.f;
#pragma unroll
    for (int w = 0; w < WAVES; ++w) s += ldsLoss[w];
    ((float*)ws)[WS_PART + blockIdx.x] = s;     // deterministic partial (no atomics)
  }
}

__global__ void vq_loss_final(const float* __restrict__ part, float* __restrict__ out) {
  __shared__ float buf[256];
  float s = 0.f;
  for (int i = threadIdx.x; i < NBLKS; i += 256) s += part[i];
  buf[threadIdx.x] = s;
  __syncthreads();
  for (int off = 128; off > 0; off >>= 1) {
    if (threadIdx.x < off) buf[threadIdx.x] += buf[threadIdx.x + off];
    __syncthreads();
  }
  if (threadIdx.x == 0) out[LOSS_OFF] = 1.25f * buf[0];  // q_loss + 0.25*e_loss
}

extern "C" void kernel_launch(void* const* d_in, const int* in_sizes, int n_in,
                              void* d_out, int out_size, void* d_ws, size_t ws_size,
                              hipStream_t stream) {
  const float* z = (const float*)d_in[0];
  const float* E = (const float*)d_in[1];
  float* out = (float*)d_out;
  unsigned int* ws = (unsigned int*)d_ws;

  vq_prep_frags<<<128, 256, 0, stream>>>(E, ws);
  vq_prep_e2 <<<4,   256, 0, stream>>>(E, (float*)(ws + WS_E2));
  vq_main    <<<NBLKS, WAVES * 32, 0, stream>>>(z, E, ws, out);
  vq_loss_final<<<1, 256, 0, stream>>>((const float*)(ws + WS_PART), out);
}
